// MultiHeadAttention_5222680232560
// MI455X (gfx1250) — matrix operand order repaired and checked by reading, not compiled, not run on hardware
//
#include <hip/hip_runtime.h>

// ---------------------------------------------------------------------------
// MI455X (gfx1250) multi-head attention, bf16 WMMA + f32 accumulate.
// B=4, S=1024, D=A=1024, H=16, HD=64.
// d_out = out [B,S,D] fp32 (4M floats) ++ weights [B,H,S,S] fp32 (64M floats)
// ---------------------------------------------------------------------------

typedef __attribute__((ext_vector_type(4)))  float  f4;
typedef __attribute__((ext_vector_type(8)))  float  v8f;
typedef __attribute__((ext_vector_type(16))) __bf16 v16bf;

#define MODE_PROJ_QK 0
#define MODE_PROJ_V  1
#define MODE_SCORES  2
#define MODE_PV      3
#define MODE_OUT     4

union Cvt8 { f4 v; __bf16 h[8]; };
union F8   { f4 v[2]; float f[8]; };

// Load one 16x32 bf16 WMMA fragment (16 bf16 per lane) from LDS as two
// 16-byte chunks. Offsets follow the CDNA5 VGPR layouts (ISA 7.12.2).
__device__ __forceinline__ v16bf ld_frag(const char* p, int o0, int o1) {
    union { v16bf v; f4 q[2]; } u;
    u.q[0] = *(const f4*)(p + o0);
    u.q[1] = *(const f4*)(p + o1);
    return u.v;
}

// ---------------------------------------------------------------------------
// LDS-tiled WMMA GEMM, 256 threads (8 waves), BK=32.
// Default tile: BM=128 x BN=128; each wave computes 32x64 (2x4 WMMA tiles).
// PV mode (N=64): BM=128 x BN=64; each wave computes 16x64.
// A row-major [M x 1024] (f32 converted on load, or bf16).
// B pre-transposed bf16 [N x 1024] (row n = column n of logical B).
// Global->LDS staging is branch-free and register-double-buffered.
// ---------------------------------------------------------------------------
template <int MODE>
__global__ __launch_bounds__(256)
void gemm_wmma(const float* __restrict__ Af, const __bf16* __restrict__ Ab,
               const __bf16* __restrict__ Bt, const float* __restrict__ bias,
               float* __restrict__ Cf, __bf16* __restrict__ Cb)
{
    constexpr int BM = 128, BK = 32;
    constexpr int BN = (MODE == MODE_PV) ? 64 : 128;
    constexpr int WM = (MODE == MODE_PV) ? 1 : 2;   // 16-row tiles per wave (M)
    constexpr int WN = 4;                            // 16-col tiles per wave (N)
    constexpr int NWCOL = BN / (16 * WN);            // waves across N
    constexpr int ASTR = 80;   // 32 bf16 (64B) + 16B pad -> conflict-free frags
    constexpr int BSTR = 80;
    constexpr bool AF32 = (MODE == MODE_PROJ_QK || MODE == MODE_PROJ_V || MODE == MODE_PV);
    constexpr int  KD   = (MODE == MODE_SCORES) ? 64 : 1024;
    constexpr int ACH = BM / 64;                     // 16B chunks per thread (A)
    constexpr int BCH = BN / 64;                     // 16B chunks per thread (B)

    __shared__ __align__(16) char sA[BM * ASTR];
    __shared__ __align__(16) char sB[BN * BSTR];

    const int t  = threadIdx.x;
    const int bx = blockIdx.x, by = blockIdx.y, bz = blockIdx.z;
    const int bb = bz >> 4, hh = bz & 15;   // batch / head for batched modes

    size_t aBase, bBase;
    if constexpr (MODE == MODE_SCORES) {
        aBase = ((size_t)(bb * 1024 + by * BM)) * 1024 + hh * 64;
        bBase = ((size_t)(bb * 1024 + bx * BN)) * 1024 + hh * 64;
    } else if constexpr (MODE == MODE_PV) {
        aBase = ((size_t)(bz * 1024 + by * BM)) * 1024;
        bBase = ((size_t)(bb * 1024 + hh * 64)) * 1024;
    } else {
        aBase = ((size_t)(by * BM)) * 1024;
        bBase = ((size_t)(bx * BN)) * 1024;
    }

    const int lane = t & 31, wv = t >> 5, half = lane >> 4, ln = lane & 15;
    const int m0 = (wv / NWCOL) * (16 * WM);
    const int n0 = (wv % NWCOL) * (16 * WN);

    // Per-thread staging coordinates (chunk = 8 elements = one 16B LDS store).
    int aRow[ACH], aOff[ACH], bRow[BCH], bOff[BCH];
#pragma unroll
    for (int i = 0; i < ACH; ++i) {
        const int c = t + i * 256;
        aRow[i] = c >> 2; aOff[i] = (c & 3) * 8;
    }
#pragma unroll
    for (int i = 0; i < BCH; ++i) {
        const int c = t + i * 256;
        bRow[i] = c >> 2; bOff[i] = (c & 3) * 8;
    }

    // Register staging buffers (double-buffer global->LDS).
    F8  raF[ACH];   // used when A is f32
    f4  raH[ACH];   // used when A is bf16
    f4  rbH[BCH];

    auto load_tile = [&](int k) {
#pragma unroll
        for (int i = 0; i < ACH; ++i) {
            if constexpr (AF32) {
                const f4* s = (const f4*)(Af + aBase + (size_t)aRow[i] * 1024 + k + aOff[i]);
                raF[i].v[0] = s[0]; raF[i].v[1] = s[1];
            } else {
                raH[i] = *(const f4*)(Ab + aBase + (size_t)aRow[i] * 1024 + k + aOff[i]);
            }
        }
#pragma unroll
        for (int i = 0; i < BCH; ++i)
            rbH[i] = *(const f4*)(Bt + bBase + (size_t)bRow[i] * 1024 + k + bOff[i]);
    };

    load_tile(0);

    v8f acc[WM][WN] = {};

    for (int k0 = 0; k0 < KD; k0 += BK) {
        __syncthreads();                 // LDS consumed by previous iteration
        // Stage registers -> LDS (with f32->bf16 conversion on the A path).
#pragma unroll
        for (int i = 0; i < ACH; ++i) {
            char* dst = sA + aRow[i] * ASTR + aOff[i] * 2;
            if constexpr (AF32) {
                Cvt8 u;
#pragma unroll
                for (int e = 0; e < 8; ++e) u.h[e] = (__bf16)raF[i].f[e];
                *(f4*)dst = u.v;
            } else {
                *(f4*)dst = raH[i];
            }
        }
#pragma unroll
        for (int i = 0; i < BCH; ++i)
            *(f4*)(sB + bRow[i] * BSTR + bOff[i] * 2) = rbH[i];

        // Issue next K-tile's global loads before the barrier so their latency
        // overlaps the barrier + WMMA compute below.
        if (k0 + BK < KD) load_tile(k0 + BK);
        __syncthreads();

        // Hoist ALL fragment loads, then issue all WMMAs back-to-back.
        v16bf afr[WM], bfr[WN];
#pragma unroll
        for (int i = 0; i < WM; ++i)
            afr[i] = ld_frag(sA + (m0 + i * 16 + ln) * ASTR, half * 16, half * 16 + 32);
#pragma unroll
        for (int j = 0; j < WN; ++j)
            bfr[j] = ld_frag(sB + (n0 + j * 16 + ln) * BSTR, half * 16, half * 16 + 32);
#pragma unroll
        for (int i = 0; i < WM; ++i)
#pragma unroll
            for (int j = 0; j < WN; ++j)
                acc[i][j] = __builtin_amdgcn_wmma_f32_16x16x32_bf16(
                    false, afr[i], false, bfr[j], (short)0, acc[i][j], false, false);
    }

    // Epilogue. C layout: VGPR r, lanes 0-15 -> (M=r, N=ln); lanes 16-31 -> (M=8+r).
#pragma unroll
    for (int i = 0; i < WM; ++i) {
        const int mBase = by * BM + m0 + i * 16 + half * 8;
#pragma unroll
        for (int j = 0; j < WN; ++j) {
            const int nl = n0 + j * 16 + ln;            // local N within block tile
            if constexpr (MODE == MODE_PROJ_QK) {
                const int n = bx * BN + nl;
                const float bsv = bias[n];
#pragma unroll
                for (int r = 0; r < 8; ++r)
                    Cb[(size_t)(mBase + r) * 1024 + n] = (__bf16)(acc[i][j][r] + bsv);
            } else if constexpr (MODE == MODE_PROJ_V) {
                const int n = bx * BN + nl;             // n = h*64 + hd
                const float bsv = bias[n];
#pragma unroll
                for (int r = 0; r < 8; ++r) {
                    const int m = mBase + r;            // m = b*1024 + s
                    const int bI = m >> 10, s = m & 1023;
                    Cb[((size_t)(bI * 1024 + n)) * 1024 + s] = (__bf16)(acc[i][j][r] + bsv);
                }
            } else if constexpr (MODE == MODE_SCORES) {
                const int n = bx * BN + nl;             // key index
#pragma unroll
                for (int r = 0; r < 8; ++r)
                    Cf[((size_t)bz * 1024 + (mBase + r)) * 1024 + n] = acc[i][j][r] * 0.125f;
            } else if constexpr (MODE == MODE_PV) {
#pragma unroll
                for (int r = 0; r < 8; ++r)
                    Cb[((size_t)(bb * 1024 + mBase + r)) * 1024 + hh * 64 + nl] =
                        (__bf16)acc[i][j][r];
            } else { // MODE_OUT
                const int n = bx * BN + nl;
                const float bsv = bias[n];
#pragma unroll
                for (int r = 0; r < 8; ++r)
                    Cf[(size_t)(mBase + r) * 1024 + n] = acc[i][j][r] + bsv;
            }
        }
    }
}

// ---------------------------------------------------------------------------
// Transpose+convert weights: W[1024][1024] f32 -> Wt[n][k] bf16.
// ---------------------------------------------------------------------------
__global__ __launch_bounds__(256)
void transpose_cvt(const float* __restrict__ src, __bf16* __restrict__ dst)
{
    __shared__ float tile[32][33];
    const int bx = blockIdx.x, by = blockIdx.y;
    const int tx = threadIdx.x, ty = threadIdx.y;   // (32, 8)
#pragma unroll
    for (int i = 0; i < 4; ++i)
        tile[ty + i * 8][tx] = src[(size_t)(by * 32 + ty + i * 8) * 1024 + bx * 32 + tx];
    __syncthreads();
#pragma unroll
    for (int i = 0; i < 4; ++i)
        dst[(size_t)(bx * 32 + ty + i * 8) * 1024 + by * 32 + tx] =
            (__bf16)tile[tx][ty + i * 8];
}

// ---------------------------------------------------------------------------
// Row softmax over 1024-wide rows, in place (fp32). 1 block per row.
// ---------------------------------------------------------------------------
__global__ __launch_bounds__(256)
void softmax_rows(float* __restrict__ w)
{
    float* p = w + (size_t)blockIdx.x * 1024;
    const int t = threadIdx.x, wv = t >> 5, ln = t & 31;
    __shared__ float red[8];

    f4 v = ((const f4*)p)[t];
    float m = fmaxf(fmaxf(v.x, v.y), fmaxf(v.z, v.w));
#pragma unroll
    for (int o = 16; o > 0; o >>= 1) m = fmaxf(m, __shfl_xor(m, o));
    if (ln == 0) red[wv] = m;
    __syncthreads();
    float mAll = red[0];
#pragma unroll
    for (int i = 1; i < 8; ++i) mAll = fmaxf(mAll, red[i]);
    __syncthreads();

    f4 e;
    e.x = __expf(v.x - mAll); e.y = __expf(v.y - mAll);
    e.z = __expf(v.z - mAll); e.w = __expf(v.w - mAll);
    float s = e.x + e.y + e.z + e.w;
#pragma unroll
    for (int o = 16; o > 0; o >>= 1) s += __shfl_xor(s, o);
    if (ln == 0) red[wv] = s;
    __syncthreads();
    float sAll = 0.f;
#pragma unroll
    for (int i = 0; i < 8; ++i) sAll += red[i];
    const float inv = 1.0f / sAll;

    e.x *= inv; e.y *= inv; e.z *= inv; e.w *= inv;
    ((f4*)p)[t] = e;
}

// ---------------------------------------------------------------------------
extern "C" void kernel_launch(void* const* d_in, const int* in_sizes, int n_in,
                              void* d_out, int out_size, void* d_ws, size_t ws_size,
                              hipStream_t stream)
{
    (void)in_sizes; (void)n_in; (void)out_size; (void)ws_size;

    const float* x  = (const float*)d_in[0];
    const float* Wq = (const float*)d_in[1];
    const float* bq = (const float*)d_in[2];
    const float* Wk = (const float*)d_in[3];
    const float* bk = (const float*)d_in[4];
    const float* Wv = (const float*)d_in[5];
    const float* bv = (const float*)d_in[6];
    const float* Wo = (const float*)d_in[7];
    const float* bo = (const float*)d_in[8];

    float* out     = (float*)d_out;
    float* weights = out + (size_t)4 * 1024 * 1024;   // [B,H,S,S]

    char* ws = (char*)d_ws;
    const size_t MB = 1024 * 1024;
    __bf16* Wqt = (__bf16*)(ws + 0 * MB);
    __bf16* Wkt = (__bf16*)(ws + 2 * MB);
    __bf16* Wvt = (__bf16*)(ws + 4 * MB);
    __bf16* Wot = (__bf16*)(ws + 6 * MB);
    __bf16* Qb  = (__bf16*)(ws + 8 * MB);     // [B,S,A] bf16
    __bf16* Kb  = (__bf16*)(ws + 16 * MB);    // [B,S,A] bf16
    __bf16* Vt  = (__bf16*)(ws + 24 * MB);    // [B,H,HD,S] bf16
    __bf16* Ctx = (__bf16*)(ws + 32 * MB);    // [B,S,A] bf16

    // 1. Weight transposes (f32 -> bf16, [K][N] -> [N][K]).
    dim3 tg(32, 32), tb(32, 8);
    transpose_cvt<<<tg, tb, 0, stream>>>(Wq, Wqt);
    transpose_cvt<<<tg, tb, 0, stream>>>(Wk, Wkt);
    transpose_cvt<<<tg, tb, 0, stream>>>(Wv, Wvt);
    transpose_cvt<<<tg, tb, 0, stream>>>(Wo, Wot);

    // 2. QKV projections (M=4096, N=1024, K=1024), 128x128 tiles.
    dim3 gProj(8, 32, 1);
    gemm_wmma<MODE_PROJ_QK><<<gProj, 256, 0, stream>>>(x, nullptr, Wqt, bq, nullptr, Qb);
    gemm_wmma<MODE_PROJ_QK><<<gProj, 256, 0, stream>>>(x, nullptr, Wkt, bk, nullptr, Kb);
    gemm_wmma<MODE_PROJ_V ><<<gProj, 256, 0, stream>>>(x, nullptr, Wvt, bv, nullptr, Vt);

    // 3. Scores = scale * Q K^T  (per (b,h): M=1024, N=1024, K=64) -> fp32 weights.
    gemm_wmma<MODE_SCORES><<<dim3(8, 8, 64), 256, 0, stream>>>(
        nullptr, Qb, Kb, nullptr, weights, nullptr);

    // 4. Softmax rows, in place.
    softmax_rows<<<dim3(64 * 1024), 256, 0, stream>>>(weights);

    // 5. ctx = softmax(scores) @ V  (per (b,h): M=1024, N=64, K=1024).
    gemm_wmma<MODE_PV><<<dim3(1, 8, 64), 256, 0, stream>>>(
        weights, nullptr, Vt, nullptr, nullptr, Ctx);

    // 6. out = ctx @ Wo + bo  (M=4096, N=1024, K=1024) -> fp32.
    gemm_wmma<MODE_OUT><<<dim3(8, 32, 1), 256, 0, stream>>>(
        nullptr, Ctx, Wot, bo, out, nullptr);
}
